// MultiHeadSelfAttn_72825465471295
// MI455X (gfx1250) — matrix operand order repaired and checked by reading, not compiled, not run on hardware
//
#include <hip/hip_runtime.h>
#include <hip/hip_bf16.h>

#define N_     8
#define CIN    256
#define T_     128
#define V_     128
#define OUT_   512
#define H_     8
#define D_     64
#define TV     (T_ * V_)          // 16384
#define NTH    (N_ * T_ * H_)     // 8192 attention tiles
#define QSIZE  ((size_t)NTH * V_ * D_)  // 67,108,864 elems per buffer

typedef __attribute__((ext_vector_type(16))) __bf16 v16bf;
typedef __attribute__((ext_vector_type(8)))  __bf16 v8bf;
typedef __attribute__((ext_vector_type(8)))  float  v8f;

union V16 { v16bf v; v8bf h[2]; };

__device__ __forceinline__ v8bf cvt8(float4 lo, float4 hi) {
    v8bf r;
    r[0] = (__bf16)lo.x; r[1] = (__bf16)lo.y; r[2] = (__bf16)lo.z; r[3] = (__bf16)lo.w;
    r[4] = (__bf16)hi.x; r[5] = (__bf16)hi.y; r[6] = (__bf16)hi.z; r[7] = (__bf16)hi.w;
    return r;
}

__device__ __forceinline__ v8f wmma_bf16(v16bf a, v16bf b, v8f c) {
    return __builtin_amdgcn_wmma_f32_16x16x32_bf16(false, a, false, b, (short)0, c, false, false);
}

// ---------------------------------------------------------------------------
// Kernel 1: Q/K/V projections.  Per-wave 16(M) x 64(cols) tile, K=256 loop.
// p=0 -> Q [n][t][h][v][d], p=1 -> K^T [n][t][h][d][v], p=2 -> V [n][t][h][v][d]
// ---------------------------------------------------------------------------
__global__ __launch_bounds__(256) void qkv_kernel(
    const float* __restrict__ x,
    const float* __restrict__ wq, const float* __restrict__ wk,
    const float* __restrict__ wv,
    __bf16* __restrict__ qb, __bf16* __restrict__ ktb, __bf16* __restrict__ vb)
{
    const int wid  = blockIdx.x * 8 + (threadIdx.x >> 5);
    const int lane = threadIdx.x & 31;
    const int ct = wid & 255;          // 256 col tiles of 64
    const int mt = (wid >> 8) & 31;    // 32 m tiles of 16
    const int n  = (wid >> 13) & 7;    // 8 batches
    const int p  = wid >> 16;          // 3 projections

    const float* W = (p == 0) ? wq : ((p == 1) ? wk : wv);
    const int laneHalf = lane >> 4, laneRow = lane & 15;
    const int mRow   = mt * 16 + laneRow;
    const int colBase = ct * 64;
    const float* xn = x + (size_t)n * CIN * TV;

    v8f acc[4] = {};
#pragma unroll
    for (int kc = 0; kc < CIN; kc += 32) {
        // A operand: 16x32 chunk of W (row = mRow; K runs per lane-half)
        V16 a;
        const float* ap = W + (size_t)mRow * CIN + kc + laneHalf * 8;
        a.h[0] = cvt8(*(const float4*)ap,        *(const float4*)(ap + 4));
        a.h[1] = cvt8(*(const float4*)(ap + 16), *(const float4*)(ap + 20));
        // B operand: lane holds column laneRow, K in table order (kc+8*laneHalf+s, +16)
        const int kk = kc + laneHalf * 8;
        const float* bp0 = xn + (size_t)kk * TV + colBase + laneRow;
#pragma unroll
        for (int nt = 0; nt < 4; ++nt) {
            const float* bpp = bp0 + nt * 16;
            V16 b;
#pragma unroll
            for (int s = 0; s < 8; ++s) {
                b.h[0][s] = (__bf16)bpp[(size_t)s * TV];
                b.h[1][s] = (__bf16)bpp[(size_t)(s + 16) * TV];
            }
            acc[nt] = wmma_bf16(a.v, b.v, acc[nt]);
        }
    }
    // C/D layout: lane holds col = lane%16 (+16*nt), row = vg + 8*(lane/16)
#pragma unroll
    for (int nt = 0; nt < 4; ++nt) {
#pragma unroll
        for (int vg = 0; vg < 8; ++vg) {
            const int o   = mt * 16 + vg + 8 * laneHalf;
            const int col = colBase + nt * 16 + laneRow;
            const int h = o >> 6, d = o & 63;
            const int t = col >> 7, v = col & 127;
            const size_t head = ((size_t)n * T_ + t) * H_ + h;
            const __bf16 val = (__bf16)acc[nt][vg];
            if (p == 1)       ktb[(head * D_ + d) * V_ + v] = val;
            else if (p == 0)  qb [(head * V_ + v) * D_ + d] = val;
            else              vb [(head * V_ + v) * D_ + d] = val;
        }
    }
}

// ---------------------------------------------------------------------------
// Kernel 2: attention per (n,t,h) tile.  8 waves; wave w owns rows [16w,16w+16)
// of S (128x128).  S = Q K^T (*sqrt(D)), softmax rows, O = P V / rowsum.
// Staging uses the CDNA5 async memory->LDS path (ASYNCcnt).
// ---------------------------------------------------------------------------
__global__ __launch_bounds__(256) void attn_kernel(
    const __bf16* __restrict__ qb, const __bf16* __restrict__ ktb,
    const __bf16* __restrict__ vb, __bf16* __restrict__ ob)
{
    __shared__ __align__(16) char smem[49152];
    __bf16* sQ  = (__bf16*)smem;              // 128x64 = 16 KB
    __bf16* sKT = (__bf16*)(smem + 16384);    // 64x128 = 16 KB
    __bf16* sV  = (__bf16*)(smem + 32768);    // 128x64 = 16 KB
    __bf16* sP  = (__bf16*)smem;              // overlays sQ/sKT after barrier

    const int bid  = blockIdx.x;              // (n*T + t)*H + h
    const size_t base = (size_t)bid * (V_ * D_);
    const int tid = threadIdx.x;

    // Stage Q, K^T, V (each a contiguous 16 KB block) into LDS
    const float4* gq = (const float4*)(qb  + base);
    const float4* gk = (const float4*)(ktb + base);
    const float4* gv = (const float4*)(vb  + base);
    float4* lq = (float4*)sQ; float4* lk = (float4*)sKT; float4* lv = (float4*)sV;
#if defined(__gfx1250__)
    // async global->LDS streaming copies, tracked on ASYNCcnt.
    // VDST holds the 32-bit LDS byte offset (low 32 bits of the shared-aperture
    // generic address); VADDR is the 64-bit global address.
#pragma unroll
    for (int i = 0; i < 4; ++i) {
        const int idx = tid + i * 256;
        unsigned lqa = (unsigned)(unsigned long long)(const void*)(lq + idx);
        unsigned lka = (unsigned)(unsigned long long)(const void*)(lk + idx);
        unsigned lva = (unsigned)(unsigned long long)(const void*)(lv + idx);
        asm volatile("global_load_async_to_lds_b128 %0, %1, off"
                     :: "v"(lqa), "v"((unsigned long long)(gq + idx)) : "memory");
        asm volatile("global_load_async_to_lds_b128 %0, %1, off"
                     :: "v"(lka), "v"((unsigned long long)(gk + idx)) : "memory");
        asm volatile("global_load_async_to_lds_b128 %0, %1, off"
                     :: "v"(lva), "v"((unsigned long long)(gv + idx)) : "memory");
    }
    asm volatile("s_wait_asynccnt 0x0" ::: "memory");
#else
#pragma unroll
    for (int i = 0; i < 4; ++i) {
        const int idx = tid + i * 256;
        lq[idx] = gq[idx]; lk[idx] = gk[idx]; lv[idx] = gv[idx];
    }
#endif
    __syncthreads();

    const int w = tid >> 5, lane = tid & 31;
    const int laneHalf = lane >> 4, laneRow = lane & 15;
    const int mBase = w * 16;

    // --- S = Q K^T over K-dim D=64 (2 chunks of 32) ---
    v8f acc[8] = {};
#pragma unroll
    for (int kc = 0; kc < D_; kc += 32) {
        V16 a;
        const int row = mBase + laneRow;
        const int off = kc + laneHalf * 8;
        a.h[0] = *(const v8bf*)(sQ + row * D_ + off);
        a.h[1] = *(const v8bf*)(sQ + row * D_ + off + 16);
        const __bf16* brow = sKT + (kc + laneHalf * 8) * V_ + laneRow;
#pragma unroll
        for (int cb = 0; cb < 8; ++cb) {
            v16bf b;
#pragma unroll
            for (int s = 0; s < 8; ++s) {
                b[s] = brow[cb * 16 + s * V_];
                b[s + 8] = brow[cb * 16 + (s + 16) * V_];
            }
            acc[cb] = wmma_bf16(a.v, b, acc[cb]);
        }
    }

    // --- row softmax (scale = *8 = sqrt(D); reference divides by D^-0.5) ---
    float inv[8];
#pragma unroll
    for (int vg = 0; vg < 8; ++vg) {
        float rm = -3.4e38f;
#pragma unroll
        for (int cb = 0; cb < 8; ++cb) rm = fmaxf(rm, acc[cb][vg] * 8.0f);
#pragma unroll
        for (int mask = 1; mask <= 8; mask <<= 1)
            rm = fmaxf(rm, __shfl_xor(rm, mask, 32));
        float rs = 0.0f;
#pragma unroll
        for (int cb = 0; cb < 8; ++cb) {
            float e = __expf(acc[cb][vg] * 8.0f - rm);
            acc[cb][vg] = e;
            rs += e;
        }
#pragma unroll
        for (int mask = 1; mask <= 8; mask <<= 1)
            rs += __shfl_xor(rs, mask, 32);
        inv[vg] = 1.0f / rs;
    }
    __syncthreads();   // all waves done reading sQ/sKT before overlaying with P

    // --- write P (bf16) to LDS: C-layout -> row-major 16x128 per wave ---
    __bf16* myP = sP + w * (16 * 128);
#pragma unroll
    for (int cb = 0; cb < 8; ++cb)
#pragma unroll
        for (int vg = 0; vg < 8; ++vg)
            myP[(vg + 8 * laneHalf) * 128 + cb * 16 + laneRow] = (__bf16)acc[cb][vg];
    __syncthreads();

    // --- O = P V over K-dim 128 (4 chunks of 32) ---
    v8f accO[4] = {};
#pragma unroll
    for (int kw = 0; kw < V_; kw += 32) {
        V16 a;
        const int off = kw + laneHalf * 8;
        a.h[0] = *(const v8bf*)(myP + laneRow * 128 + off);
        a.h[1] = *(const v8bf*)(myP + laneRow * 128 + off + 16);
        const __bf16* brow = sV + (kw + laneHalf * 8) * D_ + laneRow;
#pragma unroll
        for (int nb = 0; nb < 4; ++nb) {
            v16bf b;
#pragma unroll
            for (int s = 0; s < 8; ++s) {
                b[s] = brow[nb * 16 + s * D_];
                b[s + 8] = brow[nb * 16 + (s + 16) * D_];
            }
            accO[nb] = wmma_bf16(a.v, b, accO[nb]);
        }
    }

    // normalize rows, store transposed as [n][t][h][d][v]
#pragma unroll
    for (int nb = 0; nb < 4; ++nb)
#pragma unroll
        for (int vg = 0; vg < 8; ++vg) {
            const int vrow = mBase + vg + 8 * laneHalf;
            const int d    = nb * 16 + laneRow;
            ob[base + (size_t)d * V_ + vrow] = (__bf16)(accO[nb][vg] * inv[vg]);
        }
}

// ---------------------------------------------------------------------------
// Kernel 3: out = wp @ O + bp.  Per-wave 16(M) x 64(cols) tile, K=512 loop.
// ---------------------------------------------------------------------------
__global__ __launch_bounds__(256) void outproj_kernel(
    const float* __restrict__ wp, const float* __restrict__ bpv,
    const __bf16* __restrict__ ob, float* __restrict__ out)
{
    const int wid  = blockIdx.x * 8 + (threadIdx.x >> 5);
    const int lane = threadIdx.x & 31;
    const int ct = wid & 255;
    const int mt = (wid >> 8) & 31;
    const int n  = (wid >> 13) & 7;
    const int laneHalf = lane >> 4, laneRow = lane & 15;
    const int mRow = mt * 16 + laneRow;
    const int colBase = ct * 64;
    const __bf16* obn = ob + (size_t)n * OUT_ * TV;  // [t][h][d][v]

    v8f acc[4] = {};
#pragma unroll 4
    for (int kc = 0; kc < OUT_; kc += 32) {
        V16 a;
        const float* ap = wp + (size_t)mRow * OUT_ + kc + laneHalf * 8;
        a.h[0] = cvt8(*(const float4*)ap,        *(const float4*)(ap + 4));
        a.h[1] = cvt8(*(const float4*)(ap + 16), *(const float4*)(ap + 20));
        const int c = kc + laneHalf * 8; // lane's first K channel (table order)
#pragma unroll
        for (int nt = 0; nt < 4; ++nt) {
            const int col0 = colBase + nt * 16;
            const int t = col0 >> 7, v0 = col0 & 127;
            const __bf16* bq = obn + ((size_t)t * (H_ * D_) + c) * V_ + v0 + laneRow;
            v16bf b;
#pragma unroll
            for (int s = 0; s < 8; ++s) {
                b[s] = bq[(size_t)s * V_];
                b[s + 8] = bq[(size_t)(s + 16) * V_];
            }
            acc[nt] = wmma_bf16(a.v, b, acc[nt]);
        }
    }
#pragma unroll
    for (int nt = 0; nt < 4; ++nt)
#pragma unroll
        for (int vg = 0; vg < 8; ++vg) {
            const int m   = mt * 16 + vg + 8 * laneHalf;
            const int col = colBase + nt * 16 + laneRow;
            out[((size_t)n * OUT_ + m) * TV + col] = acc[nt][vg] + bpv[m];
        }
}

extern "C" void kernel_launch(void* const* d_in, const int* in_sizes, int n_in,
                              void* d_out, int out_size, void* d_ws, size_t ws_size,
                              hipStream_t stream) {
    const float* x  = (const float*)d_in[0];
    const float* wq = (const float*)d_in[1];
    const float* wk = (const float*)d_in[2];
    const float* wv = (const float*)d_in[3];
    const float* wp = (const float*)d_in[4];
    const float* bp = (const float*)d_in[5];
    float* out = (float*)d_out;

    __bf16* qb  = (__bf16*)d_ws;
    __bf16* ktb = qb  + QSIZE;
    __bf16* vb  = ktb + QSIZE;
    __bf16* ob  = vb  + QSIZE;

    // 3 proj * 8 n * 32 mtiles * 256 coltiles = 196608 waves / 8 per block
    qkv_kernel<<<24576, 256, 0, stream>>>(x, wq, wk, wv, qb, ktb, vb);
    // one workgroup per (n,t,h)
    attn_kernel<<<NTH, 256, 0, stream>>>(qb, ktb, vb, ob);
    // 8 n * 32 mtiles * 256 coltiles = 65536 waves / 8 per block
    outproj_kernel<<<8192, 256, 0, stream>>>(wp, bp, ob, out);
}
